// GraphMultiHeadAttention_4320737100266
// MI455X (gfx1250) — matrix operand order repaired and checked by reading, not compiled, not run on hardware
//
#include <hip/hip_runtime.h>

// ---------------------------------------------------------------------------
// GraphMultiHeadAttention for MI455X (gfx1250, wave32, WMMA bf16 path)
// B=16 S=512 D=128 H=8 DH=16 E=64 EH=8
// Dominant cost: streaming edge_attributes (1.07 GB). mean-over-EH is folded
// into the edge weight -> [.,64]x[64,8] GEMM. All GEMMs use
// v_wmma_f32_16x16x32_bf16 with f32 accumulation.
// Total HBM ~1.21 GB  =>  ~52 us at 23.3 TB/s (roofline ~46 us).
// ---------------------------------------------------------------------------

typedef __attribute__((ext_vector_type(16))) __bf16 v16bf;
typedef __attribute__((ext_vector_type(8)))  float  v8f;

#define B_  16
#define S_  512
#define D_  128
#define H_  8
#define E_  64
#define SCALE_ 0.08838834764831845f  // 128^-0.5

__device__ __forceinline__ unsigned short f2bf(float f) {
  unsigned int u = __float_as_uint(f);
  unsigned int r = (u + 0x7FFFu + ((u >> 16) & 1u)) >> 16;  // round-nearest-even
  return (unsigned short)r;
}
__device__ __forceinline__ float bf2f(unsigned short s) {
  return __uint_as_float(((unsigned int)s) << 16);
}

union ABFrag {           // one WMMA A/B operand: 16 bf16 = 8 VGPRs
  v16bf v;
  unsigned short s[16];
};

// ---------------------------------------------------------------------------
// Kernel 1: fold We head-mean into M[16][64] bf16 (rows 8..15 zero pad),
//           convert Wq/Wk/Wv/Wo to bf16.
// ---------------------------------------------------------------------------
__global__ void __launch_bounds__(256) prep_kernel(
    const float* __restrict__ Wq, const float* __restrict__ Wk,
    const float* __restrict__ Wv, const float* __restrict__ Wo,
    const float* __restrict__ We,
    unsigned short* __restrict__ Wbf,   // [4][128][128]
    unsigned short* __restrict__ Mbf) { // [16][64]
  const float* Ws[4] = {Wq, Wk, Wv, Wo};
  for (int m = 0; m < 4; ++m)
    for (int i = threadIdx.x; i < D_ * D_; i += 256)
      Wbf[m * D_ * D_ + i] = f2bf(Ws[m][i]);
  for (int i = threadIdx.x; i < 16 * E_; i += 256) {
    int h = i / E_, e = i % E_;
    float v = 0.f;
    if (h < H_) {
      for (int j = 0; j < 8; ++j) v += We[(h * 8 + j) * E_ + e];
      v *= 0.125f;  // mean over EH
    }
    Mbf[i] = f2bf(v);
  }
}

// ---------------------------------------------------------------------------
// Kernel 2: Q/K/V projections. Block = one 16-row tile of X=[8192,128],
// 8 waves, wave w == head w. Outputs head-major bf16:
//   Qh,Kh: [B,H,S,16] row-major ; Vt: [B,H,16,S] (transposed for PV B-loads)
// ---------------------------------------------------------------------------
__global__ void __launch_bounds__(256) qkv_kernel(
    const float* __restrict__ X, const unsigned short* __restrict__ Wbf,
    unsigned short* __restrict__ Qh, unsigned short* __restrict__ Kh,
    unsigned short* __restrict__ Vt) {
  __shared__ unsigned short Xs[16 * 128];
  int b  = blockIdx.x >> 5;
  int sb = (blockIdx.x & 31) << 4;
  for (int i = threadIdx.x; i < 16 * 128; i += 256) {
    int r = i >> 7, d = i & 127;
    Xs[i] = f2bf(X[((size_t)(b * S_ + sb + r)) * D_ + d]);
  }
  __syncthreads();
  int wave = threadIdx.x >> 5, lane = threadIdx.x & 31;
  int hi = lane >> 4, l15 = lane & 15;

  // A fragments (shared across the 3 weight matrices)
  ABFrag a[4];
  for (int c = 0; c < 4; ++c) {
    const unsigned short* src = &Xs[l15 * 128 + 32 * c + (hi ? 8 : 0)];
    for (int t = 0; t < 8; ++t) { a[c].s[t] = src[t]; a[c].s[8 + t] = src[16 + t]; }
  }
  for (int m = 0; m < 3; ++m) {
    const unsigned short* W = Wbf + (size_t)m * D_ * D_;
    v8f C = {};
    for (int c = 0; c < 4; ++c) {
      ABFrag bm;
      int n = wave * 16 + l15;                       // output column (head*16+dh)
      const unsigned short* wsrc = &W[n * 128 + 32 * c + (hi ? 8 : 0)];
      for (int t = 0; t < 8; ++t) { bm.s[t] = wsrc[t]; bm.s[8 + t] = wsrc[16 + t]; }
      C = __builtin_amdgcn_wmma_f32_16x16x32_bf16(false, a[c].v, false, bm.v,
                                                  (short)0, C, false, false);
    }
    for (int i = 0; i < 8; ++i) {
      int srow = sb + i + (hi ? 8 : 0);
      unsigned short val = f2bf(C[i]);
      size_t hb = (size_t)(b * H_ + wave);
      if (m == 0)      Qh[(hb * S_ + srow) * 16 + l15] = val;
      else if (m == 1) Kh[(hb * S_ + srow) * 16 + l15] = val;
      else             Vt[(hb * 16 + l15) * S_ + srow] = val;
    }
  }
}

// ---------------------------------------------------------------------------
// Kernel 3: edge bias. Block = one (b,q): streams edge[b,q,0:512,0:64] (128KB),
// bias[k,h] = edge @ M^T via 2 WMMAs (K=64) per 16-k tile; transpose through
// LDS so the global bf16 store to Ebias[B,H,S,S] is fully coalesced.
// Lane pairs (l, l+16) together cover the full 256B edge row -> full line use.
// ---------------------------------------------------------------------------
__global__ void __launch_bounds__(256) ebias_kernel(
    const float* __restrict__ Eattr, const unsigned short* __restrict__ Mbf,
    unsigned short* __restrict__ Ebias) {
  __shared__ float biasLds[H_ * S_];  // 16 KB
  int b = blockIdx.x >> 9;
  int q = blockIdx.x & 511;
  int wave = threadIdx.x >> 5, lane = threadIdx.x & 31;
  int hi = lane >> 4, l15 = lane & 15;

  ABFrag bm[2];
  for (int c = 0; c < 2; ++c) {
    const unsigned short* src = &Mbf[l15 * E_ + 32 * c + (hi ? 8 : 0)];
    for (int t = 0; t < 8; ++t) { bm[c].s[t] = src[t]; bm[c].s[8 + t] = src[16 + t]; }
  }
  const size_t rowbase = ((size_t)(b * S_ + q)) * S_;
  for (int t = wave; t < 32; t += 8) {
    int kb = t * 16;
    v8f C = {};
    for (int c = 0; c < 2; ++c) {
      ABFrag a;
      const float* src = &Eattr[(rowbase + kb + l15) * E_ + 32 * c + (hi ? 8 : 0)];
      for (int tt = 0; tt < 8; ++tt) {
        a.s[tt]     = f2bf(src[tt]);
        a.s[8 + tt] = f2bf(src[16 + tt]);
      }
      C = __builtin_amdgcn_wmma_f32_16x16x32_bf16(false, a.v, false, bm[c].v,
                                                  (short)0, C, false, false);
    }
    if (l15 < H_)  // columns 8..15 are pad
      for (int i = 0; i < 8; ++i)
        biasLds[l15 * S_ + kb + i + (hi ? 8 : 0)] = C[i];
  }
  __syncthreads();
  int h  = threadIdx.x >> 5;
  int kk = (threadIdx.x & 31) * 16;
  size_t obase = (((size_t)(b * H_ + h) * S_) + q) * S_ + kk;
  for (int t = 0; t < 16; ++t) Ebias[obase + t] = f2bf(biasLds[h * S_ + kk + t]);
}

// ---------------------------------------------------------------------------
// Kernel 4: attention. Block = 4 waves per (b,h,16-q tile).
//   scores = (Q K^T) * bias, mask -> -3e38, *SCALE  -> LDS f32 [16][512]
//     (each wave owns 8 of the 32 k-tiles)
//   softmax: 8 threads per row, tree reduce; unnormalized P in bf16
//   O = P V: each wave accumulates 4 K=32 chunks, partials reduced via LDS;
//   1/rowsum folded into the final coalesced AO store.
// ---------------------------------------------------------------------------
__global__ void __launch_bounds__(128) attn_kernel(
    const unsigned short* __restrict__ Qh, const unsigned short* __restrict__ Kh,
    const unsigned short* __restrict__ Vt, const unsigned short* __restrict__ Ebias,
    const unsigned char* __restrict__ mask, unsigned short* __restrict__ AO) {
  __shared__ float          sS[16 * 512];   // 32 KB scores
  __shared__ unsigned short sP[16 * 512];   // 16 KB probabilities (bf16)
  __shared__ float Opart[4 * 256];          // 4 KB partial P@V accumulators
  __shared__ float red[128];
  __shared__ float rowMax[16];
  __shared__ float rowRecip[16];

  int b  = blockIdx.x >> 8;
  int h  = (blockIdx.x >> 5) & 7;
  int qb = (blockIdx.x & 31) << 4;
  int tid  = threadIdx.x;
  int wave = tid >> 5, lane = tid & 31;
  int hi = lane >> 4, l15 = lane & 15;
  const size_t headBase = (size_t)(b * H_ + h) * S_;
  const size_t mbase    = (size_t)b * S_ * S_;

  // Q A-fragment, K-dim padded 16 -> 32 with zeros (same in every wave)
  ABFrag qa;
  {
    const unsigned short* src = &Qh[(headBase + qb + l15) * 16 + (hi ? 8 : 0)];
    for (int t = 0; t < 8; ++t) { qa.s[t] = src[t]; qa.s[8 + t] = 0; }
  }

  // ---- score phase: wave w handles k-tiles 8w .. 8w+7 ----
  for (int t = wave * 8; t < wave * 8 + 8; ++t) {
    int kb = t * 16;
    ABFrag kf;
    {  // contraction k = 8*hi + s (s<8) = real d; s>=8 (k 16..31) = zero pad
      const unsigned short* src = &Kh[(headBase + kb + l15) * 16 + (hi ? 8 : 0)];
      for (int tt = 0; tt < 8; ++tt) { kf.s[tt] = src[tt]; kf.s[8 + tt] = 0; }
    }
    v8f C = {};
    C = __builtin_amdgcn_wmma_f32_16x16x32_bf16(false, qa.v, false, kf.v,
                                                (short)0, C, false, false);
    int k = kb + l15;
    for (int i = 0; i < 8; ++i) {
      int qi = i + (hi ? 8 : 0);
      float bias = bf2f(Ebias[(headBase + qb + qi) * S_ + k]);
      unsigned char mby = mask[mbase + (size_t)(qb + qi) * S_ + k];
      sS[qi * 512 + k] = mby ? -3.0e38f : C[i] * bias * SCALE_;
    }
  }
  __syncthreads();

  // ---- softmax pass 1: row max (8 threads per row, 64-wide segments) ----
  {
    int row = tid >> 3, seg = tid & 7;
    const float* rp = &sS[row * 512 + seg * 64];
    float m = -3.4e38f;
    for (int k = 0; k < 64; ++k) m = fmaxf(m, rp[k]);
    red[tid] = m;
    __syncthreads();
    if (tid < 16) {
      float mm = red[tid * 8];
      for (int j = 1; j < 8; ++j) mm = fmaxf(mm, red[tid * 8 + j]);
      rowMax[tid] = mm;
    }
    __syncthreads();
  }
  // ---- softmax pass 2: exp + sum, write unnormalized P (bf16) ----
  {
    int row = tid >> 3, seg = tid & 7;
    const float* rp = &sS[row * 512 + seg * 64];
    unsigned short* pp = &sP[row * 512 + seg * 64];
    float mx = rowMax[row], sum = 0.f;
    for (int k = 0; k < 64; ++k) {
      float e = __expf(rp[k] - mx);
      pp[k] = f2bf(e);
      sum += e;
    }
    red[tid] = sum;
    __syncthreads();
    if (tid < 16) {
      float ss = 0.f;
      for (int j = 0; j < 8; ++j) ss += red[tid * 8 + j];
      rowRecip[tid] = 1.0f / ss;
    }
    __syncthreads();
  }

  // ---- O = P @ V : wave w accumulates chunks 4w .. 4w+3 (K=32 each) ----
  v8f O = {};
  const size_t vbase = (size_t)(b * H_ + h) * 16;
  for (int c = wave * 4; c < wave * 4 + 4; ++c) {
    int kc = c * 32;
    ABFrag pa, vf;
    const unsigned short* psrc = &sP[l15 * 512 + kc + (hi ? 8 : 0)];
    for (int t = 0; t < 8; ++t) { pa.s[t] = psrc[t]; pa.s[8 + t] = psrc[16 + t]; }
    const unsigned short* vsrc = &Vt[(vbase + l15) * S_ + kc + (hi ? 8 : 0)];
    for (int t = 0; t < 8; ++t) { vf.s[t] = vsrc[t]; vf.s[8 + t] = vsrc[16 + t]; }
    O = __builtin_amdgcn_wmma_f32_16x16x32_bf16(false, pa.v, false, vf.v,
                                                (short)0, O, false, false);
  }
  // stash partials: element (i,lane) -> row i + 8*hi, col l15
  for (int i = 0; i < 8; ++i)
    Opart[wave * 256 + (i + (hi ? 8 : 0)) * 16 + l15] = O[i];
  __syncthreads();
  // cooperative reduce + normalized, coalesced store (2 elements per thread)
  for (int e = tid * 2; e < tid * 2 + 2; ++e) {
    int qi = e >> 4, d = e & 15;
    float val = Opart[e] + Opart[256 + e] + Opart[512 + e] + Opart[768 + e];
    val *= rowRecip[qi];
    AO[((size_t)(b * S_ + qb + qi)) * D_ + h * 16 + d] = f2bf(val);
  }
}

// ---------------------------------------------------------------------------
// Kernel 5: out = AO @ Wo^T, f32 output.
// ---------------------------------------------------------------------------
__global__ void __launch_bounds__(256) outproj_kernel(
    const unsigned short* __restrict__ AO, const unsigned short* __restrict__ Wobf,
    float* __restrict__ Out) {
  __shared__ unsigned short Xs[16 * 128];
  int b  = blockIdx.x >> 5;
  int sb = (blockIdx.x & 31) << 4;
  for (int i = threadIdx.x; i < 16 * 128; i += 256)
    Xs[i] = AO[((size_t)(b * S_ + sb)) * D_ + i];
  __syncthreads();
  int wave = threadIdx.x >> 5, lane = threadIdx.x & 31;
  int hi = lane >> 4, l15 = lane & 15;
  v8f C = {};
  for (int c = 0; c < 4; ++c) {
    ABFrag a, bm;
    const unsigned short* src = &Xs[l15 * 128 + 32 * c + (hi ? 8 : 0)];
    for (int t = 0; t < 8; ++t) { a.s[t] = src[t]; a.s[8 + t] = src[16 + t]; }
    int n = wave * 16 + l15;
    const unsigned short* wsrc = &Wobf[n * 128 + 32 * c + (hi ? 8 : 0)];
    for (int t = 0; t < 8; ++t) { bm.s[t] = wsrc[t]; bm.s[8 + t] = wsrc[16 + t]; }
    C = __builtin_amdgcn_wmma_f32_16x16x32_bf16(false, a.v, false, bm.v,
                                                (short)0, C, false, false);
  }
  for (int i = 0; i < 8; ++i) {
    int srow = sb + i + (hi ? 8 : 0);
    Out[((size_t)(b * S_ + srow)) * D_ + wave * 16 + l15] = C[i];
  }
}

// ---------------------------------------------------------------------------
extern "C" void kernel_launch(void* const* d_in, const int* in_sizes, int n_in,
                              void* d_out, int out_size, void* d_ws, size_t ws_size,
                              hipStream_t stream) {
  (void)in_sizes; (void)n_in; (void)out_size; (void)ws_size;
  const float* queries = (const float*)d_in[0];
  const float* eattr   = (const float*)d_in[1];
  const unsigned char* mask = (const unsigned char*)d_in[2];  // numpy bool = 1 byte
  const float* Wq = (const float*)d_in[3];
  const float* Wk = (const float*)d_in[4];
  const float* Wv = (const float*)d_in[5];
  const float* We = (const float*)d_in[6];
  const float* Wo = (const float*)d_in[7];

  char* ws = (char*)d_ws;
  size_t off = 0;
  auto alloc = [&](size_t bytes) {
    void* p = ws + off;
    off = (off + bytes + 255) & ~(size_t)255;
    return p;
  };
  unsigned short* Wbf   = (unsigned short*)alloc((size_t)4 * D_ * D_ * 2);
  unsigned short* Mbf   = (unsigned short*)alloc((size_t)16 * E_ * 2);
  unsigned short* Qh    = (unsigned short*)alloc((size_t)B_ * H_ * S_ * 16 * 2);
  unsigned short* Kh    = (unsigned short*)alloc((size_t)B_ * H_ * S_ * 16 * 2);
  unsigned short* Vt    = (unsigned short*)alloc((size_t)B_ * H_ * S_ * 16 * 2);
  unsigned short* AO    = (unsigned short*)alloc((size_t)B_ * S_ * D_ * 2);
  unsigned short* Ebias = (unsigned short*)alloc((size_t)B_ * H_ * S_ * S_ * 2);

  prep_kernel<<<1, 256, 0, stream>>>(Wq, Wk, Wv, Wo, We, Wbf, Mbf);
  qkv_kernel<<<B_ * S_ / 16, 256, 0, stream>>>(queries, Wbf, Qh, Kh, Vt);
  ebias_kernel<<<B_ * S_, 256, 0, stream>>>(eattr, Mbf, Ebias);
  attn_kernel<<<B_ * H_ * (S_ / 16), 128, 0, stream>>>(Qh, Kh, Vt, Ebias, mask, AO);
  outproj_kernel<<<B_ * S_ / 16, 256, 0, stream>>>(AO, Wbf + 3 * D_ * D_, (float*)d_out);
}
